// Attention_88819923681753
// MI455X (gfx1250) — matrix operand order repaired and checked by reading, not compiled, not run on hardware
//
#include <hip/hip_runtime.h>
#include <hip/hip_bf16.h>

// ---------------------------------------------------------------------------
// Multi-head attention forward, MI455X (gfx1250), bf16 WMMA + flash softmax.
// B=4, S=2048, D=1024, H=16, HD=64.
//
// Pipeline:
//   1) cvt_f32_to_bf16      : q,k,v activations -> bf16 rows        (x3)
//   2) transpose_cvt_w      : Wq/Wk/Wv/Wo f32 [K,N] -> bf16 Wt[N,K] (x4)
//   3) gemm_bt<0>           : QKV projections, out bf16 [B,H,S,HD]  (x3)
//   4) flash_attn           : streaming softmax(QK^T)V, bf16 [B,S,D]
//   5) gemm_bt<1>           : output projection, f32 [B,S,D] + bias
// All matmuls ride v_wmma_f32_16x16x32_bf16; staging is double-buffered LDS,
// async-to-LDS (ASYNCcnt) when the toolchain exposes the builtin.
// ---------------------------------------------------------------------------

typedef __attribute__((ext_vector_type(8)))  __bf16 v8bf;
typedef __attribute__((ext_vector_type(16))) __bf16 v16bf;
typedef __attribute__((ext_vector_type(8)))  float  v8f;
typedef __attribute__((ext_vector_type(4)))  int    v4i;

union BFV { v16bf v; v8bf h[2]; };

#define B_   4
#define S_   2048
#define D_   1024
#define H_   16
#define HD_  64
#define MTOT (B_ * S_)          // 8192 rows

// ---- optional CDNA5 async-to-LDS path (GLOBAL_LOAD_ASYNC_TO_LDS_B128) -----
#if defined(__HIP_DEVICE_COMPILE__) &&                                   \
    __has_builtin(__builtin_amdgcn_global_load_async_to_lds_b128) &&     \
    __has_builtin(__builtin_amdgcn_s_wait_asynccnt)
#define HAVE_ASYNC 1
#else
#define HAVE_ASYNC 0
#endif

#if HAVE_ASYNC
typedef __attribute__((address_space(1))) v4i* gas_v4i_p;   // global int4*
typedef __attribute__((address_space(3))) v4i* las_v4i_p;   // LDS int4*

__device__ __forceinline__ void async_cp16(const void* g, void* l) {
  // copies 16 bytes global -> LDS, tracked by ASYNCcnt
  __builtin_amdgcn_global_load_async_to_lds_b128(
      (gas_v4i_p)(g), (las_v4i_p)(l), 0, 0);
}
__device__ __forceinline__ void async_wait0() {
  __builtin_amdgcn_s_wait_asynccnt(0);
}
#endif

static __device__ __forceinline__ v8f wmma_bf16(v16bf a, v16bf b, v8f c) {
  // D = A(16x32 bf16) x B(32x16 bf16) + C(16x16 f32)
  return __builtin_amdgcn_wmma_f32_16x16x32_bf16(
      /*neg_a=*/false, a, /*neg_b=*/false, b,
      /*c_mod=*/(short)0, c, /*reuse_a=*/false, /*reuse_b=*/false);
}

static __device__ __forceinline__ v8bf cvt8(const float4 f0, const float4 f1) {
  v8bf c;
  c[0] = (__bf16)f0.x; c[1] = (__bf16)f0.y; c[2] = (__bf16)f0.z; c[3] = (__bf16)f0.w;
  c[4] = (__bf16)f1.x; c[5] = (__bf16)f1.y; c[6] = (__bf16)f1.z; c[7] = (__bf16)f1.w;
  return c;
}

// ---------------------------------------------------------------------------
// Prep 1: elementwise f32 -> bf16 (8 elements / thread).
// ---------------------------------------------------------------------------
__launch_bounds__(256)
__global__ void cvt_f32_to_bf16(const float* __restrict__ in,
                                __bf16* __restrict__ out) {
  const size_t i = ((size_t)blockIdx.x * 256 + threadIdx.x) * 8;
  const float4* p = (const float4*)(in + i);
  *(v8bf*)(out + i) = cvt8(p[0], p[1]);
}

// ---------------------------------------------------------------------------
// Prep 2: W f32 [K,N] -> Wt bf16 [N,K] via LDS 64x64 tile transpose.
// ---------------------------------------------------------------------------
__launch_bounds__(256)
__global__ void transpose_cvt_w(const float* __restrict__ W,
                                __bf16* __restrict__ Wt) {
  __shared__ __bf16 tile[64][72];
  const int t  = threadIdx.x;
  const int r  = t >> 2;           // 0..63
  const int cb = (t & 3) * 16;     // 0,16,32,48
  const int n0 = blockIdx.x * 64;
  const int k0 = blockIdx.y * 64;

  const float4* wp = (const float4*)(W + (size_t)(k0 + r) * D_ + n0 + cb);
  *(v8bf*)&tile[r][cb]     = cvt8(wp[0], wp[1]);
  *(v8bf*)&tile[r][cb + 8] = cvt8(wp[2], wp[3]);
  __syncthreads();

  v8bf o0, o1;
#pragma unroll
  for (int i = 0; i < 8; ++i) o0[i] = tile[cb + i][r];
#pragma unroll
  for (int i = 0; i < 8; ++i) o1[i] = tile[cb + 8 + i][r];
  __bf16* op = Wt + (size_t)(n0 + r) * D_ + k0 + cb;
  *(v8bf*)op       = o0;
  *(v8bf*)(op + 8) = o1;
}

// ---------------------------------------------------------------------------
// Tiled GEMM: C[M,N] = A[M,K] @ Bt[N,K]^T + bias. M=8192, K=N=1024, all bf16 in.
// MODE 0: out bf16 in [B,H,S,HD] layout (QKV projection)
// MODE 1: out f32 row-major [M,N]      (output projection)
// 256 thr (8 waves), 128x128 tile, K-step 32, double-buffered LDS,
// wave grid 4(M) x 2(N): 2x4 accumulators -> 8 WMMA per K-step per wave.
// ---------------------------------------------------------------------------
template <int MODE>
__launch_bounds__(256)
__global__ void gemm_bt(const __bf16* __restrict__ A,    // [M,K] bf16
                        const __bf16* __restrict__ Btg,  // [N,K] bf16 (W^T)
                        const float*  __restrict__ bias,
                        void* __restrict__ Out) {
  constexpr int K_ = D_;
  constexpr int N_ = D_;

  __shared__ __bf16 As[2][128][40];   // pad 32->40 (80B pitch, 16B aligned)
  __shared__ __bf16 Bs[2][128][40];

  const int t    = threadIdx.x;
  const int n0   = blockIdx.x * 128;
  const int m0   = blockIdx.y * 128;
  const int wid  = t >> 5;
  const int lane = t & 31;
  const int wm   = wid & 3;
  const int wn   = wid >> 2;
  const int half = lane >> 4;
  const int l16  = lane & 15;

  const int ar = t >> 1,  ac = (t & 1) * 16;    // A-tile row / 16-elem chunk
  const int br = t & 127, bc = (t >> 7) * 16;   // B-tile row / chunk

  const __bf16* agp = A   + (size_t)(m0 + ar) * K_ + ac;
  const __bf16* bgp = Btg + (size_t)(n0 + br) * K_ + bc;

  v8f acc[2][4] = {};

  // prologue: stage k0=0 into buffer 0
  {
    const v8bf* ap = (const v8bf*)agp;
    *(v8bf*)&As[0][ar][ac]     = ap[0];
    *(v8bf*)&As[0][ar][ac + 8] = ap[1];
    const v8bf* bp = (const v8bf*)bgp;
    *(v8bf*)&Bs[0][br][bc]     = bp[0];
    *(v8bf*)&Bs[0][br][bc + 8] = bp[1];
  }
  __syncthreads();

  int p = 0;
  for (int k0 = 0; k0 < K_; k0 += 32) {
    const bool more = (k0 + 32) < K_;

    // ---- issue staging for next K-step (overlaps with WMMA below) ----
#if HAVE_ASYNC
    if (more) {
      async_cp16(agp + k0 + 32,     &As[p ^ 1][ar][ac]);
      async_cp16(agp + k0 + 40,     &As[p ^ 1][ar][ac + 8]);
      async_cp16(bgp + k0 + 32,     &Bs[p ^ 1][br][bc]);
      async_cp16(bgp + k0 + 40,     &Bs[p ^ 1][br][bc + 8]);
    }
#else
    v8bf at0, at1, bt0, bt1;
    if (more) {
      const v8bf* ap = (const v8bf*)(agp + k0 + 32);
      at0 = ap[0]; at1 = ap[1];
      const v8bf* bp = (const v8bf*)(bgp + k0 + 32);
      bt0 = bp[0]; bt1 = bp[1];
    }
#endif

    // ---- fragments + 8 WMMA on resident buffer ----
    BFV af[2], bf[4];
#pragma unroll
    for (int tm = 0; tm < 2; ++tm) {
      const int row = wm * 32 + tm * 16 + l16;              // M = l16
      af[tm].h[0] = *(const v8bf*)&As[p][row][8 * half];    // K = 8h..
      af[tm].h[1] = *(const v8bf*)&As[p][row][16 + 8 * half];
    }
#pragma unroll
    for (int tn = 0; tn < 4; ++tn) {
      const int col = wn * 64 + tn * 16 + l16;              // N = l16
      bf[tn].h[0] = *(const v8bf*)&Bs[p][col][8 * half];    // K = 8h..
      bf[tn].h[1] = *(const v8bf*)&Bs[p][col][16 + 8 * half];
    }
#pragma unroll
    for (int tm = 0; tm < 2; ++tm)
#pragma unroll
      for (int tn = 0; tn < 4; ++tn)
        acc[tm][tn] = wmma_bf16(af[tm].v, bf[tn].v, acc[tm][tn]);

    // ---- commit next buffer, single barrier, flip ----
    if (more) {
#if !HAVE_ASYNC
      *(v8bf*)&As[p ^ 1][ar][ac]     = at0;
      *(v8bf*)&As[p ^ 1][ar][ac + 8] = at1;
      *(v8bf*)&Bs[p ^ 1][br][bc]     = bt0;
      *(v8bf*)&Bs[p ^ 1][br][bc + 8] = bt1;
#else
      async_wait0();
#endif
      __syncthreads();
      p ^= 1;
    }
  }

  // ---- epilogue: + bias, store ----
#pragma unroll
  for (int tm = 0; tm < 2; ++tm) {
#pragma unroll
    for (int tn = 0; tn < 4; ++tn) {
      const int n  = n0 + wn * 64 + tn * 16 + l16;
      const float bv = bias[n];
#pragma unroll
      for (int r = 0; r < 8; ++r) {
        const int m     = m0 + wm * 32 + tm * 16 + r + 8 * half;  // C layout
        const float val = acc[tm][tn][r] + bv;
        if (MODE == 0) {
          const int b  = m >> 11;
          const int s  = m & (S_ - 1);
          const int h  = n >> 6;
          const int hd = n & (HD_ - 1);
          ((__bf16*)Out)[(((size_t)b * H_ + h) * S_ + s) * HD_ + hd] = (__bf16)val;
        } else {
          ((float*)Out)[(size_t)m * N_ + n] = val;
        }
      }
    }
  }
}

// ---------------------------------------------------------------------------
// Flash attention: softmax(Q K^T * 0.125) V per (b,h), streaming over keys.
// 256 thr (8 waves), 128 query rows / block; each wave owns 16 query rows.
// K/V tiles double-buffered; per 64-key block: 8 WMMA (QK^T) + 8 WMMA (PV).
// ---------------------------------------------------------------------------
__launch_bounds__(256)
__global__ void flash_attn(const __bf16* __restrict__ Q,
                           const __bf16* __restrict__ Kg,
                           const __bf16* __restrict__ Vg,
                           __bf16* __restrict__ O) {
  __shared__ __bf16 Ks[2][64][72];     // K block row-major  (= B^T for QK^T)
  __shared__ __bf16 Vt[2][64][72];     // V block transposed (= B^T for P V)
  __shared__ __bf16 Ps[8][16][72];     // per-wave P staging (C -> A layout)

  const int t    = threadIdx.x;
  const int wid  = t >> 5;
  const int lane = t & 31;
  const int half = lane >> 4;
  const int l16  = lane & 15;
  const int bh   = blockIdx.y;
  const int b    = bh >> 4;
  const int h    = bh & 15;
  const int qbase = blockIdx.x * 128;

  const __bf16* Qp = Q  + (size_t)bh * S_ * HD_;
  const __bf16* Kp = Kg + (size_t)bh * S_ * HD_;
  const __bf16* Vp = Vg + (size_t)bh * S_ * HD_;

  // Loop-invariant Q fragments straight from global in A-frag shape.
  BFV qf[2];
  {
    const __bf16* qr = Qp + (size_t)(qbase + wid * 16 + l16) * HD_;
#pragma unroll
    for (int ks = 0; ks < 2; ++ks) {
      qf[ks].h[0] = *(const v8bf*)(qr + ks * 32 + 8 * half);
      qf[ks].h[1] = *(const v8bf*)(qr + ks * 32 + 16 + 8 * half);
    }
  }

  float mrow[8], lrow[8];
  v8f oacc[4] = {};
#pragma unroll
  for (int r = 0; r < 8; ++r) { mrow[r] = -3.0e38f; lrow[r] = 0.f; }

  const int cr = t >> 2;           // 0..63 key row for cooperative staging
  const int cc = (t & 3) * 16;     // 0,16,32,48 hd column base

  // prologue: stage key-block 0 into buffer 0
  {
    const v8bf* kr = (const v8bf*)(Kp + (size_t)cr * HD_ + cc);
    *(v8bf*)&Ks[0][cr][cc]     = kr[0];
    *(v8bf*)&Ks[0][cr][cc + 8] = kr[1];
    const v8bf* vr = (const v8bf*)(Vp + (size_t)cr * HD_ + cc);
    const v8bf v0 = vr[0], v1 = vr[1];
#pragma unroll
    for (int i = 0; i < 8; ++i) Vt[0][cc + i][cr]     = v0[i];
#pragma unroll
    for (int i = 0; i < 8; ++i) Vt[0][cc + 8 + i][cr] = v1[i];
  }
  __syncthreads();

  int p = 0;
  for (int kb = 0; kb < S_; kb += 64) {
    const bool more = (kb + 64) < S_;

    // ---- issue staging of next K/V block (overlaps with compute) ----
    v8bf nv0, nv1;
#if !HAVE_ASYNC
    v8bf nk0, nk1;
#endif
    if (more) {
      const __bf16* kn = Kp + (size_t)(kb + 64 + cr) * HD_ + cc;
#if HAVE_ASYNC
      async_cp16(kn,     &Ks[p ^ 1][cr][cc]);
      async_cp16(kn + 8, &Ks[p ^ 1][cr][cc + 8]);
#else
      nk0 = ((const v8bf*)kn)[0];
      nk1 = ((const v8bf*)kn)[1];
#endif
      const __bf16* vn = Vp + (size_t)(kb + 64 + cr) * HD_ + cc;
      nv0 = ((const v8bf*)vn)[0];
      nv1 = ((const v8bf*)vn)[1];
    }

    // ---- S = Q K^T (16 x 64), K-dim = HD = 64 ----
    v8f sc[4] = {};
#pragma unroll
    for (int ks = 0; ks < 2; ++ks) {
#pragma unroll
      for (int nt = 0; nt < 4; ++nt) {
        BFV bfr;
        const int row = nt * 16 + l16;                       // key = N
        bfr.h[0] = *(const v8bf*)&Ks[p][row][ks * 32 + 8 * half];
        bfr.h[1] = *(const v8bf*)&Ks[p][row][ks * 32 + 16 + 8 * half];
        sc[nt] = wmma_bf16(qf[ks].v, bfr.v, sc[nt]);
      }
    }

    // ---- online softmax (rows in 16-lane halves; xor<16 stays inside) ----
    float alpha[8];
#pragma unroll
    for (int r = 0; r < 8; ++r) {
      float mx = -3.0e38f;
#pragma unroll
      for (int nt = 0; nt < 4; ++nt) {
        sc[nt][r] *= 0.125f;                                 // HD^-0.5
        mx = fmaxf(mx, sc[nt][r]);
      }
      mx = fmaxf(mx, __shfl_xor(mx, 1, 32));
      mx = fmaxf(mx, __shfl_xor(mx, 2, 32));
      mx = fmaxf(mx, __shfl_xor(mx, 4, 32));
      mx = fmaxf(mx, __shfl_xor(mx, 8, 32));
      const float mn = fmaxf(mrow[r], mx);
      alpha[r] = __expf(mrow[r] - mn);
      mrow[r]  = mn;
      float rs = 0.f;
#pragma unroll
      for (int nt = 0; nt < 4; ++nt) {
        const float pe = __expf(sc[nt][r] - mn);
        sc[nt][r] = pe;
        rs += pe;
      }
      rs += __shfl_xor(rs, 1, 32);
      rs += __shfl_xor(rs, 2, 32);
      rs += __shfl_xor(rs, 4, 32);
      rs += __shfl_xor(rs, 8, 32);
      lrow[r] = lrow[r] * alpha[r] + rs;
    }
#pragma unroll
    for (int ht = 0; ht < 4; ++ht)
#pragma unroll
      for (int r = 0; r < 8; ++r) oacc[ht][r] *= alpha[r];

    // ---- P: C-layout regs -> row-major LDS (private per wave) ----
#pragma unroll
    for (int nt = 0; nt < 4; ++nt)
#pragma unroll
      for (int r = 0; r < 8; ++r)
        Ps[wid][r + 8 * half][nt * 16 + l16] = (__bf16)sc[nt][r];

    // ---- O += P @ V, K-dim = 64 keys (two 32-wide steps) ----
#pragma unroll
    for (int ks = 0; ks < 2; ++ks) {
      BFV pf;
      pf.h[0] = *(const v8bf*)&Ps[wid][l16][ks * 32 + 8 * half];
      pf.h[1] = *(const v8bf*)&Ps[wid][l16][ks * 32 + 16 + 8 * half];
#pragma unroll
      for (int ht = 0; ht < 4; ++ht) {
        BFV bfr;
        const int hd = ht * 16 + l16;                        // hd = N
        bfr.h[0] = *(const v8bf*)&Vt[p][hd][ks * 32 + 8 * half];
        bfr.h[1] = *(const v8bf*)&Vt[p][hd][ks * 32 + 16 + 8 * half];
        oacc[ht] = wmma_bf16(pf.v, bfr.v, oacc[ht]);
      }
    }

    // ---- commit next buffer, single barrier, flip ----
    if (more) {
#if !HAVE_ASYNC
      *(v8bf*)&Ks[p ^ 1][cr][cc]     = nk0;
      *(v8bf*)&Ks[p ^ 1][cr][cc + 8] = nk1;
#endif
#pragma unroll
      for (int i = 0; i < 8; ++i) Vt[p ^ 1][cc + i][cr]     = nv0[i];
#pragma unroll
      for (int i = 0; i < 8; ++i) Vt[p ^ 1][cc + 8 + i][cr] = nv1[i];
#if HAVE_ASYNC
      async_wait0();
#endif
      __syncthreads();
      p ^= 1;
    }
  }

  // ---- normalize, store to [B,S,D] bf16 (merged heads) ----
#pragma unroll
  for (int ht = 0; ht < 4; ++ht) {
#pragma unroll
    for (int r = 0; r < 8; ++r) {
      const int srow = qbase + wid * 16 + r + 8 * half;
      const int col  = h * HD_ + ht * 16 + l16;
      O[((size_t)b * S_ + srow) * D_ + col] = (__bf16)(oacc[ht][r] / lrow[r]);
    }
  }
}

// ---------------------------------------------------------------------------
extern "C" void kernel_launch(void* const* d_in, const int* in_sizes, int n_in,
                              void* d_out, int out_size, void* d_ws, size_t ws_size,
                              hipStream_t stream) {
  (void)in_sizes; (void)n_in; (void)out_size; (void)ws_size;

  const float* q  = (const float*)d_in[0];
  const float* k  = (const float*)d_in[1];
  const float* v  = (const float*)d_in[2];
  const float* Wq = (const float*)d_in[3];
  const float* bq = (const float*)d_in[4];
  const float* Wk = (const float*)d_in[5];
  const float* bk = (const float*)d_in[6];
  const float* Wv = (const float*)d_in[7];
  const float* bv = (const float*)d_in[8];
  const float* Wo = (const float*)d_in[9];
  const float* bo = (const float*)d_in[10];

  // Workspace (bf16), NE = 8192*1024 elements:
  //   [0]      Qb   [B,H,S,HD]
  //   [NE]     Kb
  //   [2*NE]   Vb
  //   [3*NE]   Xk   (bf16 activations)
  //   [4*NE]   Xv
  //   [5*NE]   Xq / Ob  (aliased: Xq consumed before flash writes Ob)
  //   [6*NE]   Wqt,Wkt,Wvt,Wot  (4 x D*D)
  // Total: (6*8M + 4*1M) * 2B = 104 MB.
  const size_t NE = (size_t)MTOT * D_;
  const size_t NW = (size_t)D_ * D_;
  __bf16* Qb  = (__bf16*)d_ws;
  __bf16* Kb  = Qb + NE;
  __bf16* Vb  = Kb + NE;
  __bf16* Xk  = Vb + NE;
  __bf16* Xv  = Xk + NE;
  __bf16* Xq  = Xv + NE;
  __bf16* Ob  = Xq;               // alias (safe: stream-ordered)
  __bf16* Wqt = Xq + NE;
  __bf16* Wkt = Wqt + NW;
  __bf16* Wvt = Wkt + NW;
  __bf16* Wot = Wvt + NW;

  const dim3 blk(256);
  const dim3 gcvt((unsigned)(NE / (256 * 8)));      // 4096
  const dim3 gtr(D_ / 64, D_ / 64);                 // 16 x 16
  const dim3 gg(D_ / 128, MTOT / 128);              // 8 x 64
  const dim3 gfa(S_ / 128, B_ * H_);                // 16 x 64

  cvt_f32_to_bf16<<<gcvt, blk, 0, stream>>>(q, Xq);
  cvt_f32_to_bf16<<<gcvt, blk, 0, stream>>>(k, Xk);
  cvt_f32_to_bf16<<<gcvt, blk, 0, stream>>>(v, Xv);

  transpose_cvt_w<<<gtr, blk, 0, stream>>>(Wq, Wqt);
  transpose_cvt_w<<<gtr, blk, 0, stream>>>(Wk, Wkt);
  transpose_cvt_w<<<gtr, blk, 0, stream>>>(Wv, Wvt);
  transpose_cvt_w<<<gtr, blk, 0, stream>>>(Wo, Wot);

  gemm_bt<0><<<gg, blk, 0, stream>>>(Xq, Wqt, bq, Qb);
  gemm_bt<0><<<gg, blk, 0, stream>>>(Xk, Wkt, bk, Kb);
  gemm_bt<0><<<gg, blk, 0, stream>>>(Xv, Wvt, bv, Vb);

  flash_attn<<<gfa, blk, 0, stream>>>(Qb, Kb, Vb, Ob);

  gemm_bt<1><<<gg, blk, 0, stream>>>(Ob, Wot, bo, d_out);
}
